// TempAggregation_43585328120070
// MI455X (gfx1250) — matrix operand order repaired and checked by reading, not compiled, not run on hardware
//
#include <hip/hip_runtime.h>
#include <math.h>

// ---------------------------------------------------------------------------
// Autoformer-style temporal aggregation, fused for MI455X (gfx1250, wave32).
//
// Kernel 1 (per block: 16 rows x 1 head):
//   - 4 bf16 WMMA GEMMs (qtw, qh, kh, vh), K=128, LDS-staged
//     (activation tile via global_load_async_to_lds, ASYNCcnt)
//   - time encoding fused in the epilogue (cos+sin)
//   - circular feature-dim autocorrelation (== fft*conj -> irfft) in LDS
//   - top-4 + softmax + circular-shift aggregation -> agg (d_ws)
//   - corr transposed in LDS, attn emitted via global_store_async_from_lds
// Kernel 2 (per block: 16 rows x all 128 cols):
//   - bf16 WMMA FC GEMM (K=1024, both tiles async-loaded to LDS)
//     + bias + residual + LayerNorm -> out
// ---------------------------------------------------------------------------

#define B_ 16
#define L_ 1024
#define D_ 128
#define H_ 8
#define M_TOT (B_ * L_)

typedef __attribute__((ext_vector_type(16))) __bf16 v16bf;
typedef __attribute__((ext_vector_type(8)))  float  v8f;

static __device__ __forceinline__ v8f wmma_bf16(v16bf a, v16bf b, v8f c) {
  // (neg_a, A, neg_b, B, c_mod, C, reuse_a, reuse_b)
  return __builtin_amdgcn_wmma_f32_16x16x32_bf16(false, a, false, b, (short)0, c,
                                                 false, false);
}

// --- CDNA5 async global<->LDS copies (ASYNCcnt); inline asm is the
// --- toolchain-portable path per the bridge doc. Generic pointers to LDS
// --- carry the byte offset in their low 32 bits (aperture in [63:32]).
static __device__ __forceinline__ unsigned lds_off(const void* p) {
  return (unsigned)(unsigned long long)p;
}
static __device__ __forceinline__ void async_load_b64(unsigned lds, const void* g) {
  asm volatile("global_load_async_to_lds_b64 %0, %1, off"
               :: "v"(lds), "v"((unsigned long long)g) : "memory");
}
static __device__ __forceinline__ void async_load_b128x4(unsigned lds, const void* g) {
  asm volatile("global_load_async_to_lds_b128 %0, %1, off\n\t"
               "global_load_async_to_lds_b128 %0, %1, off offset:16\n\t"
               "global_load_async_to_lds_b128 %0, %1, off offset:32\n\t"
               "global_load_async_to_lds_b128 %0, %1, off offset:48"
               :: "v"(lds), "v"((unsigned long long)g) : "memory");
}
static __device__ __forceinline__ void async_store_b128x2(void* g, unsigned lds) {
  asm volatile("global_store_async_from_lds_b128 %0, %1, off\n\t"
               "global_store_async_from_lds_b128 %0, %1, off offset:16"
               :: "v"((unsigned long long)g), "v"(lds) : "memory");
}
static __device__ __forceinline__ void wait_async0() {
  asm volatile("s_wait_asynccnt 0x0" ::: "memory");
}

// ===========================================================================
// Kernel 1: fused projections + time enc + autocorrelation + top-k aggregation
// grid: (H, M_TOT/16), block: 256 threads = 8 waves (wave w owns cols w*16..+15)
// ===========================================================================
__global__ __launch_bounds__(256) void
proj_corr_agg_kernel(const float* __restrict__ q, const float* __restrict__ k,
                     const float* __restrict__ v,
                     const float* __restrict__ q_time, const float* __restrict__ k_time,
                     const float* __restrict__ W_qtw, const float* __restrict__ W_q,
                     const float* __restrict__ W_k, const float* __restrict__ W_v,
                     const float* __restrict__ te_w,
                     float* __restrict__ agg, float* __restrict__ attn)
{
  __shared__ float  sAf[16][32];    // activation tile (f32, async-filled)
  __shared__ __bf16 sW[128][32];    // weight tile for this head (128 cols x 32 k)
  __shared__ __bf16 sW2[128][32];   // W_qtw tile (q phase only)
  __shared__ float  sQ[16][128];    // qh rows (time-encoded)
  __shared__ float  sK[16][128];    // kh rows (time-encoded)
  __shared__ float  sV[16][128];    // vh rows
  __shared__ float  sCT[128][17];   // corr, TRANSPOSED [d][row] (+pad)

  const int h    = blockIdx.x;
  const int m0   = blockIdx.y * 16;        // 16 consecutive (b,l) rows, same b
  const int t    = threadIdx.x;
  const int lane = t & 31;
  const int wv   = t >> 5;
  const int mrow = lane & 15;              // WMMA A-operand row
  const int half = lane >> 4;              // lane group (0:lanes 0-15, 1:16-31)
  const int ncol = wv * 16 + (lane & 15);  // output column d (0..127) of this wave

  // ---- three projection passes: p=0 -> qh(+qtw), p=1 -> kh, p=2 -> vh ----
  for (int p = 0; p < 3; ++p) {
    const float* xin   = (p == 0) ? q   : (p == 1) ? k   : v;
    const float* Wmain = (p == 0) ? W_q : (p == 1) ? W_k : W_v;

    v8f acc  = {0.f, 0.f, 0.f, 0.f, 0.f, 0.f, 0.f, 0.f};
    v8f acc2 = {0.f, 0.f, 0.f, 0.f, 0.f, 0.f, 0.f, 0.f};

    #pragma unroll
    for (int s = 0; s < 4; ++s) {           // K = 128 = 4 x 32
      const int k0 = s * 32;
      __syncthreads();                       // LDS WAR vs previous step's reads
      { // activations: 16 rows x 32 k f32, async DMA, 8B per thread
        const int idx = t * 2;
        const int r = idx >> 5, c = idx & 31;
        async_load_b64(lds_off(&sAf[r][c]),
                       xin + (size_t)(m0 + r) * D_ + k0 + c);
      }
      { // weights: 128 rows x 32 k, f32 -> bf16 via VGPRs (L2-resident)
        const int n = t >> 1, c0 = (t & 1) * 16;
        const float* wp = Wmain + (size_t)(h * D_ + n) * D_ + k0 + c0;
        #pragma unroll
        for (int i = 0; i < 16; ++i) sW[n][c0 + i] = (__bf16)wp[i];
        if (p == 0) {
          const float* wp2 = W_qtw + (size_t)(h * D_ + n) * D_ + k0 + c0;
          #pragma unroll
          for (int i = 0; i < 16; ++i) sW2[n][c0 + i] = (__bf16)wp2[i];
        }
        if (s < 3) __builtin_prefetch(wp + 32, 0, 3);   // next k-slice
      }
      wait_async0();
      __syncthreads();
      // build wave32 WMMA operands per the 16-bit A / B VGPR layouts
      v16bf a, bmat;
      #pragma unroll
      for (int i = 0; i < 8; ++i) a[i]     = (__bf16)sAf[mrow][half * 8 + i];
      #pragma unroll
      for (int i = 0; i < 8; ++i) a[8 + i] = (__bf16)sAf[mrow][16 + half * 8 + i];
      #pragma unroll
      for (int i = 0; i < 16; ++i) bmat[i] = sW[ncol][half * 8 + i + (i >= 8 ? 8 : 0)];
      acc = wmma_bf16(a, bmat, acc);
      if (p == 0) {
        v16bf b2;
        #pragma unroll
        for (int i = 0; i < 16; ++i) b2[i] = sW2[ncol][half * 8 + i + (i >= 8 ? 8 : 0)];
        acc2 = wmma_bf16(a, b2, acc2);
      }
    }

    // ---- epilogue: fuse time encoding, park full rows in LDS ----
    // C layout: acc[r] belongs to local row (r + half*8), column ncol.
    const int d = ncol;
    #pragma unroll
    for (int r2 = 0; r2 < 8; ++r2) {
      const int row = r2 + half * 8;
      const int m = m0 + row;
      float val = acc[r2];
      if (p == 0) {
        const float ph = te_w[d] * q_time[m];
        val += acc2[r2] * (__cosf(ph) + __sinf(ph));
        sQ[row][d] = val;
      } else if (p == 1) {
        const float ph = te_w[d] * k_time[m];
        val += (__cosf(ph) + __sinf(ph));
        sK[row][d] = val;
      } else {
        sV[row][d] = val;
      }
    }
  }
  __syncthreads();

  // ---- circular autocorrelation + top-4 + softmax + aggregation ----
  const int bI = m0 >> 10;           // batch index (L == 1024)
  const int l0 = m0 & (L_ - 1);
  #pragma unroll
  for (int rr = 0; rr < 2; ++rr) {   // each wave owns 2 rows
    const int row = wv * 2 + rr;
    float cr[4] = {0.f, 0.f, 0.f, 0.f};   // lane owns d = lane + 32*j
    for (int e = 0; e < 128; ++e) {
      const float ke = sK[row][e];         // broadcast LDS read
      #pragma unroll
      for (int j = 0; j < 4; ++j)          // conflict-free (consecutive addrs)
        cr[j] = fmaf(sQ[row][(e + lane + 32 * j) & 127], ke, cr[j]);
    }
    #pragma unroll
    for (int j = 0; j < 4; ++j) sCT[lane + 32 * j][row] = cr[j];  // transposed

    // top-4 (value, index) with jax tie-break (lower index wins)
    float cand[4];
    #pragma unroll
    for (int j = 0; j < 4; ++j) cand[j] = cr[j];
    float tw[4]; int td[4];
    #pragma unroll
    for (int ts = 0; ts < 4; ++ts) {
      float bv = cand[0]; int bj = 0;
      #pragma unroll
      for (int j = 1; j < 4; ++j) if (cand[j] > bv) { bv = cand[j]; bj = j; }
      int bd = lane + 32 * bj;
      #pragma unroll
      for (int off = 16; off > 0; off >>= 1) {   // wave32 butterfly argmax
        const float ov = __shfl_xor(bv, off, 32);
        const int   od = __shfl_xor(bd, off, 32);
        if (ov > bv || (ov == bv && od < bd)) { bv = ov; bd = od; }
      }
      tw[ts] = bv; td[ts] = bd;
      #pragma unroll
      for (int j = 0; j < 4; ++j) if (lane + 32 * j == bd) cand[j] = -3.4e38f;
    }
    // softmax over the 4 picks (tw[0] is the max by construction)
    float p4[4]; float sum = 0.f;
    #pragma unroll
    for (int ts = 0; ts < 4; ++ts) { p4[ts] = __expf(tw[ts] - tw[0]); sum += p4[ts]; }
    const float inv = 1.0f / sum;

    const int m = m0 + row;
    const int l = m & (L_ - 1);
    float* aggp = agg + (((size_t)bI * H_ + h) * L_ + l) * D_;
    #pragma unroll
    for (int j = 0; j < 4; ++j) {
      const int d = lane + 32 * j;
      float s = 0.f;
      #pragma unroll
      for (int ts = 0; ts < 4; ++ts) s += p4[ts] * sV[row][(d + td[ts]) & 127];
      aggp[d] = s * inv;
    }
  }
  __syncthreads();

  // ---- attn = corr.transpose(0,3,1,2): [B, D, H, L] ----
  // sCT row d holds 16 consecutive l values -> 64B congruent chunks, emitted
  // straight from LDS via async DMA (no VGPR round trip). S_ENDPGM's implicit
  // wait-idle covers completion.
  {
    const int d    = t >> 1;
    const int part = (t & 1) * 8;
    float* gp = attn + (((size_t)bI * D_ + d) * H_ + h) * L_ + l0 + part;
    async_store_b128x2(gp, lds_off(&sCT[d][part]));
  }
}

// ===========================================================================
// Kernel 2: out = LN(agg_flat @ W_fc^T + b_fc + residual)
// grid: M_TOT/16 blocks; block covers 16 rows x all 128 cols (8 waves x 16)
// so LayerNorm reduces entirely in-block (deterministic, no atomics).
// ===========================================================================
__global__ __launch_bounds__(256) void
fc_ln_kernel(const float* __restrict__ agg, const float* __restrict__ W_fc,
             const float* __restrict__ b_fc, const float* __restrict__ resid,
             const float* __restrict__ ln_g, const float* __restrict__ ln_b,
             float* __restrict__ out)
{
  __shared__ float sAf[16][32];    // agg tile (f32, async-filled)
  __shared__ float sBf[128][32];   // W_fc tile (f32, async-filled)
  __shared__ float psum[8][16];
  __shared__ float psq[8][16];
  __shared__ float smu[16];
  __shared__ float srv[16];

  const int m0   = blockIdx.x * 16;
  const int t    = threadIdx.x;
  const int lane = t & 31;
  const int wv   = t >> 5;
  const int mrow = lane & 15;
  const int half = lane >> 4;
  const int n    = wv * 16 + (lane & 15);    // output column 0..127

  v8f acc = {0.f, 0.f, 0.f, 0.f, 0.f, 0.f, 0.f, 0.f};

  for (int s = 0; s < 32; ++s) {             // K = 1024 = 32 x 32
    const int k0 = s * 32;
    const int hh = k0 >> 7;                  // head of this k-slice
    const int d0 = k0 & 127;                 // feature offset within head
    __syncthreads();
    { // A tile: gather agg [B,H,L,D] -> flat [m, h*128+d]; async DMA, 8B/thread
      const int idx = t * 2;
      const int r = idx >> 5, c = idx & 31;
      const int m = m0 + r; const int b = m >> 10; const int l = m & (L_ - 1);
      async_load_b64(lds_off(&sAf[r][c]),
                     agg + (((size_t)b * H_ + hh) * L_ + l) * D_ + d0 + c);
    }
    { // B tile: W_fc [128, 1024] row-major; async DMA, 64B/thread
      const int nn = t >> 1, c0 = (t & 1) * 16;
      const float* wp = W_fc + (size_t)nn * (H_ * D_) + k0 + c0;
      async_load_b128x4(lds_off(&sBf[nn][c0]), wp);
      if (s < 31) __builtin_prefetch(wp + 32, 0, 3);
    }
    wait_async0();
    __syncthreads();
    v16bf a, bm;
    #pragma unroll
    for (int i = 0; i < 8; ++i) a[i]     = (__bf16)sAf[mrow][half * 8 + i];
    #pragma unroll
    for (int i = 0; i < 8; ++i) a[8 + i] = (__bf16)sAf[mrow][16 + half * 8 + i];
    #pragma unroll
    for (int i = 0; i < 16; ++i) bm[i]   = (__bf16)sBf[n][half * 8 + i + (i >= 8 ? 8 : 0)];
    acc = wmma_bf16(a, bm, acc);
  }

  // bias + residual, then LayerNorm over the 128 columns of each row
  float vals[8], r1[8], r2v[8];
  const float bn = b_fc[n];
  #pragma unroll
  for (int r = 0; r < 8; ++r) {
    const int row = r + half * 8;
    const int m = m0 + row;
    const float vv = acc[r] + bn + resid[(size_t)m * D_ + n];
    vals[r] = vv; r1[r] = vv; r2v[r] = vv * vv;
  }
  #pragma unroll
  for (int off = 1; off < 16; off <<= 1) {   // xor<16 stays inside half-wave
    #pragma unroll
    for (int r = 0; r < 8; ++r) {
      r1[r]  += __shfl_xor(r1[r],  off, 32);
      r2v[r] += __shfl_xor(r2v[r], off, 32);
    }
  }
  if ((lane & 15) == 0) {
    #pragma unroll
    for (int r = 0; r < 8; ++r) {
      psum[wv][r + half * 8] = r1[r];
      psq[wv][r + half * 8]  = r2v[r];
    }
  }
  __syncthreads();
  if (t < 16) {
    float a1 = 0.f, a2 = 0.f;
    #pragma unroll
    for (int w = 0; w < 8; ++w) { a1 += psum[w][t]; a2 += psq[w][t]; }
    const float mu  = a1 * (1.0f / 128.0f);
    const float var = a2 * (1.0f / 128.0f) - mu * mu;
    smu[t] = mu;
    srv[t] = rsqrtf(var + 1e-6f);
  }
  __syncthreads();
  const float g = ln_g[n], bb = ln_b[n];
  #pragma unroll
  for (int r = 0; r < 8; ++r) {
    const int row = r + half * 8;
    const int m = m0 + row;
    out[(size_t)m * D_ + n] = (vals[r] - smu[row]) * srv[row] * g + bb;
  }
}

// ===========================================================================
extern "C" void kernel_launch(void* const* d_in, const int* in_sizes, int n_in,
                              void* d_out, int out_size, void* d_ws, size_t ws_size,
                              hipStream_t stream) {
  const float* q      = (const float*)d_in[0];
  const float* k      = (const float*)d_in[1];
  const float* v      = (const float*)d_in[2];
  const float* q_time = (const float*)d_in[3];
  const float* k_time = (const float*)d_in[4];
  const float* W_qtw  = (const float*)d_in[5];
  const float* W_q    = (const float*)d_in[6];
  const float* W_k    = (const float*)d_in[7];
  const float* W_v    = (const float*)d_in[8];
  const float* W_fc   = (const float*)d_in[9];
  const float* b_fc   = (const float*)d_in[10];
  const float* te_w   = (const float*)d_in[11];
  const float* ln_g   = (const float*)d_in[12];
  const float* ln_b   = (const float*)d_in[13];

  float* out  = (float*)d_out;
  float* attn = out + (size_t)B_ * L_ * D_;       // tuple output #2, [B,D,H,L]
  float* agg  = (float*)d_ws;                     // [B,H,L,D] f32 = 64 MiB

  dim3 g1(H_, M_TOT / 16, 1);
  proj_corr_agg_kernel<<<g1, 256, 0, stream>>>(q, k, v, q_time, k_time,
                                               W_qtw, W_q, W_k, W_v, te_w,
                                               agg, attn);

  dim3 g2(M_TOT / 16, 1, 1);
  fc_ln_kernel<<<g2, 256, 0, stream>>>(agg, W_fc, b_fc, q, ln_g, ln_b, out);
}
